// Qwen3MoeExperts_75892072120692
// MI455X (gfx1250) — compile-verified, matrix operand order checked by reading, not run on hardware
//
#include <hip/hip_runtime.h>

typedef __bf16 bf16;
typedef __attribute__((ext_vector_type(16))) __bf16 v16bf;
typedef __attribute__((ext_vector_type(8)))  __bf16 v8bf;
typedef __attribute__((ext_vector_type(2)))  __bf16 v2bf;
typedef __attribute__((ext_vector_type(8)))  float  v8f;

#define E_ 32
#define T_ 512
#define H_ 2048
#define I_ 768

#define BM 128          // T-tile
#define BN 128          // N-tile (I for kernel1, H for kernel2)
#define BK 64           // K step == two bf16 WMMA K-slices
#define PITCH 72        // BK + 8 pad (bf16): 144B rows -> all 16 bank-groups hit once per half-wave

union Frag { v16bf v; v8bf h[2]; };

__device__ __forceinline__ unsigned pack2(float a, float b) {
#if defined(__has_builtin) && __has_builtin(__builtin_amdgcn_cvt_pk_bf16_f32)
    union { v2bf v; unsigned u; } p;
    p.v = __builtin_amdgcn_cvt_pk_bf16_f32(a, b);   // v_cvt_pk_bf16_f32
    return p.u;
#else
    union { bf16 h[2]; unsigned u; } p;
    p.h[0] = (bf16)a; p.h[1] = (bf16)b;
    return p.u;
#endif
}

__device__ __forceinline__ float fast_rcp(float x) {
#if defined(__has_builtin) && __has_builtin(__builtin_amdgcn_rcpf)
    return __builtin_amdgcn_rcpf(x);                // v_rcp_f32
#else
    return 1.0f / x;
#endif
}

// 16x32 bf16 fragment from LDS tile, ISA 16-bit A/B layout:
// lane L holds row (L&15); K-halves 0..7/16..23 (L<16) or 8..15/24..31 (L>=16).
__device__ __forceinline__ void load_frag(const bf16* base, int rowBase, int kOff, int lane, Frag& f) {
    const int row = rowBase + (lane & 15);
    const int kb  = kOff + ((lane < 16) ? 0 : 8);
    f.h[0] = *(const v8bf*)(base + row * PITCH + kb);
    f.h[1] = *(const v8bf*)(base + row * PITCH + kb + 16);
}

#define WMMA_BF16(A, B, C) \
    __builtin_amdgcn_wmma_f32_16x16x32_bf16(false, (A), false, (B), (short)0, (C), false, false)

// ---------------------------------------------------------------------------
// Kernel 1: hidden[e,t,i] = silu(X·Wg^T) * (X·Wu^T)   (bf16 out to workspace)
// block tile 128x128xK, 8 waves as 2(M)x4(N), wave tile 64x32, double-buffered LDS
// ---------------------------------------------------------------------------
__global__ __launch_bounds__(256)
void moe_gateup_kernel(const float* __restrict__ X,    // [E,T,H]
                       const float* __restrict__ Wg,   // [E,I,H]
                       const float* __restrict__ Wu,   // [E,I,H]
                       bf16* __restrict__ Hid)         // [E,T,I]
{
    __shared__ bf16 sX[2 * BM * PITCH];
    __shared__ bf16 sG[2 * BN * PITCH];
    __shared__ bf16 sU[2 * BN * PITCH];

    const int e  = blockIdx.x;
    const int tB = blockIdx.y * BM;
    const int iB = blockIdx.z * BN;

    const int tid  = threadIdx.x;
    const int lane = tid & 31;
    const int wave = tid >> 5;
    const int wm   = (wave & 1) * 64;   // 2 waves along M -> 64 rows each
    const int wn   = (wave >> 1) * 32;  // 4 waves along N -> 32 cols each

    const float* Xb  = X  + ((size_t)e * T_ + tB) * H_;
    const float* Wgb = Wg + ((size_t)e * I_ + iB) * H_;
    const float* Wub = Wu + ((size_t)e * I_ + iB) * H_;

    v8f accG[4][2], accU[4][2];
    const v8f vzero = {0.f,0.f,0.f,0.f,0.f,0.f,0.f,0.f};
#pragma unroll
    for (int mf = 0; mf < 4; ++mf)
#pragma unroll
        for (int nf = 0; nf < 2; ++nf) { accG[mf][nf] = vzero; accU[mf][nf] = vzero; }

    // stage-in: each thread owns 32 contiguous floats of one row, for each matrix
    const int sr = tid >> 1, sc = (tid & 1) * 32;

    unsigned stX[16], stG[16], stU[16];

    auto fetch = [&](int k0) {
        const float* xp = Xb  + (size_t)sr * H_ + k0 + sc;
        const float* gp = Wgb + (size_t)sr * H_ + k0 + sc;
        const float* up = Wub + (size_t)sr * H_ + k0 + sc;
#pragma unroll
        for (int j = 0; j < 16; ++j) {
            stX[j] = pack2(xp[2*j], xp[2*j + 1]);
            stG[j] = pack2(gp[2*j], gp[2*j + 1]);
            stU[j] = pack2(up[2*j], up[2*j + 1]);
        }
    };

    auto spill = [&](int b) {
        unsigned* dX = (unsigned*)&sX[b * BM * PITCH + sr * PITCH + sc];
        unsigned* dG = (unsigned*)&sG[b * BN * PITCH + sr * PITCH + sc];
        unsigned* dU = (unsigned*)&sU[b * BN * PITCH + sr * PITCH + sc];
#pragma unroll
        for (int j = 0; j < 16; ++j) { dX[j] = stX[j]; dG[j] = stG[j]; dU[j] = stU[j]; }
    };

    auto compute = [&](int b) {
        const bf16* pX = &sX[b * BM * PITCH];
        const bf16* pG = &sG[b * BN * PITCH];
        const bf16* pU = &sU[b * BN * PITCH];
#pragma unroll
        for (int kk = 0; kk < 2; ++kk) {
            Frag a[4], bg[2], bu[2];
#pragma unroll
            for (int mf = 0; mf < 4; ++mf) load_frag(pX, wm + mf * 16, kk * 32, lane, a[mf]);
#pragma unroll
            for (int nf = 0; nf < 2; ++nf) {
                load_frag(pG, wn + nf * 16, kk * 32, lane, bg[nf]);
                load_frag(pU, wn + nf * 16, kk * 32, lane, bu[nf]);
            }
#pragma unroll
            for (int mf = 0; mf < 4; ++mf)
#pragma unroll
                for (int nf = 0; nf < 2; ++nf) {
                    accG[mf][nf] = WMMA_BF16(a[mf].v, bg[nf].v, accG[mf][nf]);
                    accU[mf][nf] = WMMA_BF16(a[mf].v, bu[nf].v, accU[mf][nf]);
                }
        }
    };

    fetch(0);
    spill(0);
    __syncthreads();

    int buf = 0;
    for (int k0 = 0; k0 < H_; k0 += BK) {
        if (k0 + BK < H_) {
            fetch(k0 + BK);      // global loads for next stage (overlap with WMMA)
            compute(buf);
            spill(buf ^ 1);      // write other buffer while peers finish reading this one
        } else {
            compute(buf);
        }
        __syncthreads();
        buf ^= 1;
    }

    // epilogue: silu(g)*u, bf16 store. D layout: vgpr r, lane L -> M = (L<16?r:8+r), N = L&15
#pragma unroll
    for (int mf = 0; mf < 4; ++mf)
#pragma unroll
        for (int nf = 0; nf < 2; ++nf) {
            const int m0 = tB + wm + mf * 16 + ((lane < 16) ? 0 : 8);
            const int n  = iB + wn + nf * 16 + (lane & 15);
            bf16* hp = Hid + ((size_t)e * T_ + m0) * I_ + n;
#pragma unroll
            for (int r = 0; r < 8; ++r) {
                const float g = accG[mf][nf][r];
                const float u = accU[mf][nf][r];
                const float s = g * u * fast_rcp(1.0f + __expf(-g));
                hp[(size_t)r * I_] = (bf16)s;
            }
        }
}

// ---------------------------------------------------------------------------
// Kernel 2: out[e,t,h] = hidden[e,t,:] · Wd[e,h,:]   (fp32 out)
// ---------------------------------------------------------------------------
__global__ __launch_bounds__(256)
void moe_down_kernel(const bf16*  __restrict__ Hid,   // [E,T,I] bf16
                     const float* __restrict__ Wd,    // [E,H,I]
                     float* __restrict__ Out)         // [E*T, H]
{
    __shared__ bf16 sA[2 * BM * PITCH];
    __shared__ bf16 sB[2 * BN * PITCH];

    const int e  = blockIdx.x;
    const int tB = blockIdx.y * BM;
    const int hB = blockIdx.z * BN;

    const int tid  = threadIdx.x;
    const int lane = tid & 31;
    const int wave = tid >> 5;
    const int wm   = (wave & 1) * 64;
    const int wn   = (wave >> 1) * 32;

    const bf16*  Ab = Hid + ((size_t)e * T_ + tB) * I_;
    const float* Bb = Wd  + ((size_t)e * H_ + hB) * I_;

    v8f acc[4][2];
    const v8f vzero = {0.f,0.f,0.f,0.f,0.f,0.f,0.f,0.f};
#pragma unroll
    for (int mf = 0; mf < 4; ++mf)
#pragma unroll
        for (int nf = 0; nf < 2; ++nf) acc[mf][nf] = vzero;

    const int sr = tid >> 1, sc = (tid & 1) * 32;  // 32 elements of one row per thread

    uint4 stA[4];        // 32 bf16 (copied raw)
    unsigned stB[16];    // 32 fp32 -> 16 packed bf16x2

    auto fetch = [&](int k0) {
        const uint4* ap = (const uint4*)(Ab + (size_t)sr * I_ + k0 + sc);
#pragma unroll
        for (int j = 0; j < 4; ++j) stA[j] = ap[j];
        const float* bp = Bb + (size_t)sr * I_ + k0 + sc;
#pragma unroll
        for (int j = 0; j < 16; ++j) stB[j] = pack2(bp[2*j], bp[2*j + 1]);
    };

    auto spill = [&](int b) {
        uint4* dA = (uint4*)&sA[b * BM * PITCH + sr * PITCH + sc];
#pragma unroll
        for (int j = 0; j < 4; ++j) dA[j] = stA[j];
        unsigned* dB = (unsigned*)&sB[b * BN * PITCH + sr * PITCH + sc];
#pragma unroll
        for (int j = 0; j < 16; ++j) dB[j] = stB[j];
    };

    auto compute = [&](int b) {
        const bf16* pA = &sA[b * BM * PITCH];
        const bf16* pB = &sB[b * BN * PITCH];
#pragma unroll
        for (int kk = 0; kk < 2; ++kk) {
            Frag a[4], bb[2];
#pragma unroll
            for (int mf = 0; mf < 4; ++mf) load_frag(pA, wm + mf * 16, kk * 32, lane, a[mf]);
#pragma unroll
            for (int nf = 0; nf < 2; ++nf) load_frag(pB, wn + nf * 16, kk * 32, lane, bb[nf]);
#pragma unroll
            for (int mf = 0; mf < 4; ++mf)
#pragma unroll
                for (int nf = 0; nf < 2; ++nf)
                    acc[mf][nf] = WMMA_BF16(a[mf].v, bb[nf].v, acc[mf][nf]);
        }
    };

    fetch(0);
    spill(0);
    __syncthreads();

    int buf = 0;
    for (int k0 = 0; k0 < I_; k0 += BK) {
        if (k0 + BK < I_) {
            fetch(k0 + BK);
            compute(buf);
            spill(buf ^ 1);
        } else {
            compute(buf);
        }
        __syncthreads();
        buf ^= 1;
    }

#pragma unroll
    for (int mf = 0; mf < 4; ++mf)
#pragma unroll
        for (int nf = 0; nf < 2; ++nf) {
            const int m0 = tB + wm + mf * 16 + ((lane < 16) ? 0 : 8);
            const int n  = hB + wn + nf * 16 + (lane & 15);
            float* op = Out + ((size_t)e * T_ + m0) * H_ + n;
#pragma unroll
            for (int r = 0; r < 8; ++r)
                op[(size_t)r * H_] = acc[mf][nf][r];
        }
}

extern "C" void kernel_launch(void* const* d_in, const int* in_sizes, int n_in,
                              void* d_out, int out_size, void* d_ws, size_t ws_size,
                              hipStream_t stream) {
    (void)in_sizes; (void)n_in; (void)out_size; (void)ws_size;
    const float* X  = (const float*)d_in[0];   // routed_input [E,T,H]
    const float* Wg = (const float*)d_in[1];   // gate_proj   [E,I,H]
    const float* Wu = (const float*)d_in[2];   // up_proj     [E,I,H]
    const float* Wd = (const float*)d_in[3];   // down_proj   [E,H,I]
    float* out = (float*)d_out;                // [E*T, H] fp32
    bf16*  hid = (bf16*)d_ws;                  // [E,T,I] bf16 scratch (24 MiB)

    dim3 g1(E_, T_ / BM, I_ / BN);             // (32, 4, 6)
    moe_gateup_kernel<<<g1, 256, 0, stream>>>(X, Wg, Wu, hid);

    dim3 g2(E_, T_ / BM, H_ / BN);             // (32, 4, 16)
    moe_down_kernel<<<g2, 256, 0, stream>>>(hid, Wd, out);
}
